// SegmentPrediction_85366769975822
// MI455X (gfx1250) — compile-verified, matrix operand order checked by reading, not run on hardware
//
#include <hip/hip_runtime.h>
#include <hip/hip_bf16.h>
#include <math.h>

// ---------------------------------------------------------------------------
// Problem constants (match reference)
// ---------------------------------------------------------------------------
#define N_NODES 100000
#define N_EDGES 400000
#define N_SEG   20000
#define A_DIM   16
#define IN_DIM  32
#define H_DIM   64
#define XF_DIM  (7 * H_DIM)   // 448
#define NNZ_SS  160000

typedef __attribute__((ext_vector_type(16))) _Float16 v16h;
typedef __attribute__((ext_vector_type(8)))  float    v8f;

// ---------------------------------------------------------------------------
// Device helpers
// ---------------------------------------------------------------------------
__device__ __forceinline__ float gelu_exact(float x) {
  return 0.5f * x * (1.0f + erff(x * 0.70710678118654752440f));
}

// monotone float <-> uint mapping for atomicMax-based segment max
__device__ __forceinline__ unsigned f32_ord(float f) {
  unsigned u = __float_as_uint(f);
  return (u & 0x80000000u) ? ~u : (u | 0x80000000u);
}
__device__ __forceinline__ float ord_f32(unsigned u) {
  return (u & 0x80000000u) ? __uint_as_float(u ^ 0x80000000u)
                           : __uint_as_float(~u);
}

__device__ __forceinline__ void cvt8(v16h& a, float4 q0, float4 q1, int base) {
  a[base + 0] = (_Float16)q0.x; a[base + 1] = (_Float16)q0.y;
  a[base + 2] = (_Float16)q0.z; a[base + 3] = (_Float16)q0.w;
  a[base + 4] = (_Float16)q1.x; a[base + 5] = (_Float16)q1.y;
  a[base + 6] = (_Float16)q1.z; a[base + 7] = (_Float16)q1.w;
}

// ---------------------------------------------------------------------------
// Weight packing: rearrange B[K,64] (f32, row-major) into the per-lane WMMA
// B-fragment layout (f16), so the GEMM reads one contiguous v16h per K-chunk.
// Layout: out[((ct*nk + kc)*32 + lane)*16 + e], where the lane holds column
// n = ct*16 + (lane&15) and e maps to k = kc*32 + koff + (e<8 ? e : e+8),
// koff = (lane&16) ? 8 : 0.  k >= K is zero-padded (K=16 case).
// ---------------------------------------------------------------------------
__global__ void pack_b(const float* __restrict__ B, int K, int nk,
                       _Float16* __restrict__ out) {
  const int total = 4 * nk * 32 * 16;
  const int idx = blockIdx.x * blockDim.x + threadIdx.x;
  if (idx >= total) return;
  const int e    = idx & 15;
  const int lane = (idx >> 4) & 31;
  const int t    = idx >> 9;          // ct*nk + kc
  const int kc   = t % nk;
  const int ct   = t / nk;
  const int koff = (lane & 16) ? 8 : 0;
  const int k    = kc * 32 + koff + (e < 8 ? e : e + 8);
  const int n    = ct * 16 + (lane & 15);
  out[idx] = (k < K) ? (_Float16)B[(size_t)k * H_DIM + n] : (_Float16)0.0f;
}

// ---------------------------------------------------------------------------
// GEMM: C[M,64] = act(A[M,K] @ B[K,64] + bias).
// One wave computes a full 16-row x 64-col strip: the A fragment is loaded
// once per K-chunk and reused by 4 WMMAs (4 col tiles), quartering A traffic.
// NK = number of 32-wide K chunks (compile-time, fully unrolled).
// TAIL = true when K = NK*32-16 (only the K=16 attr projection).
// ---------------------------------------------------------------------------
template <int NK, bool TAIL>
__global__ void wmma_gemm(const float* __restrict__ A,
                          const _Float16* __restrict__ Bp,
                          const float* __restrict__ bias,
                          float* __restrict__ C, int M, int act) {
  constexpr int KREAL = NK * 32 - (TAIL ? 16 : 0);
  const int wavesPerBlk = blockDim.x >> 5;
  const int strip  = blockIdx.x * wavesPerBlk + (threadIdx.x >> 5);
  const int strips = M >> 4;
  if (strip >= strips) return;                        // uniform per wave
  const int row0 = strip << 4;
  const int lane = threadIdx.x & 31;
  const int m    = row0 + (lane & 15);
  const int koff = (lane & 16) ? 8 : 0;
  const float*    Ar = A  + (size_t)m * KREAL + koff;
  const _Float16* Bl = Bp + (size_t)lane * 16;

  v8f acc0 = {}, acc1 = {}, acc2 = {}, acc3 = {};
#pragma unroll
  for (int kc = 0; kc < NK; ++kc) {
    const float* ap = Ar + kc * 32;
    v16h a = {};
    cvt8(a, *(const float4*)(ap), *(const float4*)(ap + 4), 0);
    if (!(TAIL && kc == NK - 1)) {   // compile-time: folds away
      cvt8(a, *(const float4*)(ap + 16), *(const float4*)(ap + 20), 8);
    }
    const v16h b0 = *(const v16h*)(Bl + (size_t)(0 * NK + kc) * 512);
    const v16h b1 = *(const v16h*)(Bl + (size_t)(1 * NK + kc) * 512);
    const v16h b2 = *(const v16h*)(Bl + (size_t)(2 * NK + kc) * 512);
    const v16h b3 = *(const v16h*)(Bl + (size_t)(3 * NK + kc) * 512);
    acc0 = __builtin_amdgcn_wmma_f32_16x16x32_f16(false, a, false, b0, (short)0, acc0, false, false);
    acc1 = __builtin_amdgcn_wmma_f32_16x16x32_f16(false, a, false, b1, (short)0, acc1, false, false);
    acc2 = __builtin_amdgcn_wmma_f32_16x16x32_f16(false, a, false, b2, (short)0, acc2, false, false);
    acc3 = __builtin_amdgcn_wmma_f32_16x16x32_f16(false, a, false, b3, (short)0, acc3, false, false);
  }

  const int nl = lane & 15;
  const int mb = row0 + ((lane & 16) ? 8 : 0);
  float bv0 = 0.0f, bv1 = 0.0f, bv2 = 0.0f, bv3 = 0.0f;
  if (bias) {
    bv0 = bias[nl]; bv1 = bias[nl + 16]; bv2 = bias[nl + 32]; bv3 = bias[nl + 48];
  }
#pragma unroll
  for (int r = 0; r < 8; ++r) {
    float* Crow = C + (size_t)(mb + r) * H_DIM + nl;
    float v0 = acc0[r] + bv0, v1 = acc1[r] + bv1;
    float v2 = acc2[r] + bv2, v3 = acc3[r] + bv3;
    if (act == 1) {
      v0 = gelu_exact(v0); v1 = gelu_exact(v1);
      v2 = gelu_exact(v2); v3 = gelu_exact(v3);
    }
    Crow[0] = v0; Crow[16] = v1; Crow[32] = v2; Crow[48] = v3;
  }
}

// ---------------------------------------------------------------------------
// GATv2 pieces
// ---------------------------------------------------------------------------
__global__ void gat_edge_logits(const float* __restrict__ xl,
                                const float* __restrict__ xr,
                                const float* __restrict__ ep,
                                const float* __restrict__ att,
                                const int* __restrict__ src,
                                const int* __restrict__ dst,
                                float* __restrict__ logit,
                                unsigned* __restrict__ mx, int nnz) {
  const int e = blockIdx.x * blockDim.x + threadIdx.x;
  if (e >= nnz) return;
  const float4* pl = (const float4*)(xl + (size_t)src[e] * H_DIM);
  const float4* pr = (const float4*)(xr + (size_t)dst[e] * H_DIM);
  const float4* pe = (const float4*)(ep + (size_t)e * H_DIM);
  const float4* pa = (const float4*)att;
  float s = 0.0f;
#pragma unroll
  for (int j = 0; j < H_DIM / 4; ++j) {
    const float4 a = pl[j], b = pr[j], c = pe[j], w = pa[j];
    float m;
    m = a.x + b.x + c.x; m = (m > 0.0f) ? m : 0.2f * m; s += m * w.x;
    m = a.y + b.y + c.y; m = (m > 0.0f) ? m : 0.2f * m; s += m * w.y;
    m = a.z + b.z + c.z; m = (m > 0.0f) ? m : 0.2f * m; s += m * w.z;
    m = a.w + b.w + c.w; m = (m > 0.0f) ? m : 0.2f * m; s += m * w.w;
  }
  logit[e] = s;
  atomicMax(mx + dst[e], f32_ord(s));
}

__global__ void gat_edge_exp(const float* __restrict__ logit,
                             const unsigned* __restrict__ mx,
                             const int* __restrict__ dst,
                             float* __restrict__ ex,
                             float* __restrict__ den, int nnz) {
  const int e = blockIdx.x * blockDim.x + threadIdx.x;
  if (e >= nnz) return;
  const int d = dst[e];
  const float v = expf(logit[e] - ord_f32(mx[d]));
  ex[e] = v;
  atomicAdd(den + d, v);
}

__global__ void gat_edge_aggregate(const float* __restrict__ ex,
                                   const float* __restrict__ den,
                                   const int* __restrict__ src,
                                   const int* __restrict__ dst,
                                   const float* __restrict__ xl,
                                   float* __restrict__ agg, int nnz) {
  const int idx = blockIdx.x * blockDim.x + threadIdx.x;
  if (idx >= nnz * H_DIM) return;
  const int e = idx >> 6;
  const int j = idx & (H_DIM - 1);
  const int d = dst[e];
  const float alpha = ex[e] / (den[d] + 1e-16f);
  atomicAdd(agg + (size_t)d * H_DIM + j,
            alpha * xl[(size_t)src[e] * H_DIM + j]);
}

// out = gelu(agg + bias) [+ pre]
__global__ void node_update(const float* __restrict__ agg,
                            const float* __restrict__ bias,
                            const float* __restrict__ pre,
                            float* __restrict__ out, int n) {
  const int idx = blockIdx.x * blockDim.x + threadIdx.x;
  if (idx >= n * H_DIM) return;
  const int j = idx & (H_DIM - 1);
  float v = gelu_exact(agg[idx] + bias[j]);
  if (pre) v += pre[idx];
  out[idx] = v;
}

// ---------------------------------------------------------------------------
// COO SpMM scatter: out[row[i]*ldo + j] += val[i] * M[col[i]*H + j]
// ---------------------------------------------------------------------------
__global__ void spmm_scatter(const int* __restrict__ row,
                             const int* __restrict__ col,
                             const float* __restrict__ val,
                             const float* __restrict__ Md,
                             float* __restrict__ out, int ldo, int nnz) {
  const int idx = blockIdx.x * blockDim.x + threadIdx.x;
  if (idx >= nnz * H_DIM) return;
  const int i = idx >> 6;
  const int j = idx & (H_DIM - 1);
  atomicAdd(out + (size_t)row[i] * ldo + j,
            val[i] * Md[(size_t)col[i] * H_DIM + j]);
}

// xf[s, 0:64]=segment_embed, [64:320]=0 (spmm targets), [320:384]=time, [384:448]=week
__global__ void xf_init(const float* __restrict__ seg_embed,
                        const float* __restrict__ time_embed,
                        const float* __restrict__ week_embed,
                        const int* __restrict__ cur_t,
                        const int* __restrict__ cur_w,
                        float* __restrict__ xf, int S) {
  const int idx = blockIdx.x * blockDim.x + threadIdx.x;
  if (idx >= S * XF_DIM) return;
  const int s = idx / XF_DIM;
  const int c = idx - s * XF_DIM;
  float v;
  if (c < 64)       v = seg_embed[(size_t)s * H_DIM + c];
  else if (c < 320) v = 0.0f;
  else if (c < 384) v = time_embed[(size_t)cur_t[0] * H_DIM + (c - 320)];
  else              v = week_embed[(size_t)cur_w[0] * H_DIM + (c - 384)];
  xf[idx] = v;
}

// ep3[e, j] = ss_val[e] * c3_We[j]   (outer product, edge_dim = 1)
__global__ void ep3_outer(const float* __restrict__ ssv,
                          const float* __restrict__ We,
                          float* __restrict__ ep, int nnz) {
  const int idx = blockIdx.x * blockDim.x + threadIdx.x;
  if (idx >= nnz * H_DIM) return;
  ep[idx] = ssv[idx >> 6] * We[idx & (H_DIM - 1)];
}

// out[s] = sigmoid([y2[s] | xf_gnn[s]] @ out_W + out_b) * 3600
__global__ void head_out(const float* __restrict__ y,
                         const float* __restrict__ g,
                         const float* __restrict__ W,
                         const float* __restrict__ b,
                         float* __restrict__ out, int S) {
  const int s = blockIdx.x * blockDim.x + threadIdx.x;
  if (s >= S) return;
  const float4* py = (const float4*)(y + (size_t)s * H_DIM);
  const float4* pg = (const float4*)(g + (size_t)s * H_DIM);
  const float4* w0 = (const float4*)W;
  const float4* w1 = (const float4*)(W + H_DIM);
  float acc = b[0];
#pragma unroll
  for (int j = 0; j < H_DIM / 4; ++j) {
    const float4 a = py[j], c = pg[j], u = w0[j], v = w1[j];
    acc += a.x * u.x + a.y * u.y + a.z * u.z + a.w * u.w;
    acc += c.x * v.x + c.y * v.y + c.z * v.z + c.w * v.w;
  }
  out[s] = 3600.0f / (1.0f + expf(-acc));
}

// ---------------------------------------------------------------------------
// Host side
// ---------------------------------------------------------------------------
static inline char* carve_bytes(char*& p, size_t bytes) {
  char* r = p;
  p += (bytes + 255) & ~(size_t)255;
  return r;
}
static inline float* carve(char*& p, size_t nfloats) {
  return (float*)carve_bytes(p, nfloats * sizeof(float));
}

// Pack one weight matrix [K,64] into f16 fragment layout (carved from ws).
static _Float16* pack_weight(char*& p, const float* B, int K, hipStream_t s) {
  const int nk = (K + 31) / 32;
  const size_t total = (size_t)4 * nk * 32 * 16;
  _Float16* out = (_Float16*)carve_bytes(p, total * sizeof(_Float16));
  pack_b<<<(int)((total + 255) / 256), 256, 0, s>>>(B, K, nk, out);
  return out;
}

template <int NK, bool TAIL>
static void launch_gemm_t(const float* A, const _Float16* Bp, const float* bias,
                          float* C, int M, int act, hipStream_t s) {
  const int strips = M / 16;
  const int wavesPerBlk = 4;                 // 128 threads
  const int blocks = (strips + wavesPerBlk - 1) / wavesPerBlk;
  wmma_gemm<NK, TAIL><<<blocks, wavesPerBlk * 32, 0, s>>>(A, Bp, bias, C, M, act);
}
static void gemm_k16(const float* A, const _Float16* Bp, const float* bias,
                     float* C, int M, int act, hipStream_t s) {
  launch_gemm_t<1, true>(A, Bp, bias, C, M, act, s);
}
static void gemm_k32(const float* A, const _Float16* Bp, const float* bias,
                     float* C, int M, int act, hipStream_t s) {
  launch_gemm_t<1, false>(A, Bp, bias, C, M, act, s);
}
static void gemm_k64(const float* A, const _Float16* Bp, const float* bias,
                     float* C, int M, int act, hipStream_t s) {
  launch_gemm_t<2, false>(A, Bp, bias, C, M, act, s);
}
static void gemm_k448(const float* A, const _Float16* Bp, const float* bias,
                      float* C, int M, int act, hipStream_t s) {
  launch_gemm_t<14, false>(A, Bp, bias, C, M, act, s);
}

static void run_gat_softmax_agg(const float* xl, const float* xr, const float* ep,
                                const float* att, const int* src, const int* dst,
                                float* logit, float* ex, unsigned* mx, float* den,
                                float* agg, int nnz, int nseg, hipStream_t s) {
  hipMemsetAsync(mx,  0, (size_t)nseg * sizeof(unsigned), s);
  hipMemsetAsync(den, 0, (size_t)nseg * sizeof(float), s);
  hipMemsetAsync(agg, 0, (size_t)nseg * H_DIM * sizeof(float), s);
  gat_edge_logits<<<(nnz + 255) / 256, 256, 0, s>>>(xl, xr, ep, att, src, dst, logit, mx, nnz);
  gat_edge_exp<<<(nnz + 255) / 256, 256, 0, s>>>(logit, mx, dst, ex, den, nnz);
  gat_edge_aggregate<<<(nnz * H_DIM + 255) / 256, 256, 0, s>>>(ex, den, src, dst, xl, agg, nnz);
}

extern "C" void kernel_launch(void* const* d_in, const int* in_sizes, int n_in,
                              void* d_out, int out_size, void* d_ws, size_t ws_size,
                              hipStream_t stream) {
  // ---- inputs (setup_inputs() dict order) ----
  const float* x_rec      = (const float*)d_in[0];
  const float* attr       = (const float*)d_in[1];
  const float* se_val     = (const float*)d_in[2];
  const float* sn_val     = (const float*)d_in[3];
  const float* ss_val     = (const float*)d_in[4];
  const float* node_embed = (const float*)d_in[5];
  const float* edge_embed = (const float*)d_in[6];
  const float* seg_embed  = (const float*)d_in[7];
  const float* time_embed = (const float*)d_in[8];
  const float* week_embed = (const float*)d_in[9];
  const float* node_lin_W = (const float*)d_in[10];
  const float* node_lin_b = (const float*)d_in[11];
  const float* attr1_W    = (const float*)d_in[12];
  const float* attr1_b    = (const float*)d_in[13];
  const float* attr2_W    = (const float*)d_in[14];
  const float* attr2_b    = (const float*)d_in[15];
  const float* c_Wl[2]    = {(const float*)d_in[16], (const float*)d_in[23]};
  const float* c_bl[2]    = {(const float*)d_in[17], (const float*)d_in[24]};
  const float* c_Wr[2]    = {(const float*)d_in[18], (const float*)d_in[25]};
  const float* c_br[2]    = {(const float*)d_in[19], (const float*)d_in[26]};
  const float* c_We[2]    = {(const float*)d_in[20], (const float*)d_in[27]};
  const float* c_att[2]   = {(const float*)d_in[21], (const float*)d_in[28]};
  const float* c_bias[2]  = {(const float*)d_in[22], (const float*)d_in[29]};
  const float* c3_Wl      = (const float*)d_in[30];
  const float* c3_bl      = (const float*)d_in[31];
  const float* c3_Wr      = (const float*)d_in[32];
  const float* c3_br      = (const float*)d_in[33];
  const float* c3_We      = (const float*)d_in[34];
  const float* c3_att     = (const float*)d_in[35];
  const float* c3_bias    = (const float*)d_in[36];
  const float* lin1_W     = (const float*)d_in[37];
  const float* lin1_b     = (const float*)d_in[38];
  const float* lin2_W     = (const float*)d_in[39];
  const float* lin2_b     = (const float*)d_in[40];
  const float* out_W      = (const float*)d_in[41];
  const float* out_b      = (const float*)d_in[42];
  const int*   edge_index = (const int*)d_in[43];
  const int*   se_row     = (const int*)d_in[44];
  const int*   se_col     = (const int*)d_in[45];
  const int*   sn_row     = (const int*)d_in[46];
  const int*   sn_col     = (const int*)d_in[47];
  const int*   ss_row     = (const int*)d_in[48];
  const int*   ss_col     = (const int*)d_in[49];
  const int*   cur_t      = (const int*)d_in[50];
  const int*   cur_w      = (const int*)d_in[51];
  float* out = (float*)d_out;
  (void)in_sizes; (void)n_in; (void)out_size; (void)ws_size;

  const int* src = edge_index;             // edge_index[0]
  const int* dst = edge_index + N_EDGES;   // edge_index[1]

  // ---- workspace carving ----
  char* p = (char*)d_ws;
  float* attr_f = carve(p, (size_t)N_EDGES * H_DIM);   // attr @ attr1 (+b)
  float* ep     = carve(p, (size_t)N_EDGES * H_DIM);   // reused: per-layer ep, attr2p, ep3
  float* hA     = carve(p, (size_t)N_NODES * H_DIM);
  float* hB     = carve(p, (size_t)N_NODES * H_DIM);
  float* nf0    = carve(p, (size_t)N_NODES * H_DIM);
  float* nfA    = carve(p, (size_t)N_NODES * H_DIM);
  float* nfB    = carve(p, (size_t)N_NODES * H_DIM);
  float* xl     = carve(p, (size_t)N_NODES * H_DIM);
  float* xr     = carve(p, (size_t)N_NODES * H_DIM);
  float* agg    = carve(p, (size_t)N_NODES * H_DIM);
  float* logit  = carve(p, (size_t)N_EDGES);
  float* ex     = carve(p, (size_t)N_EDGES);
  float* den    = carve(p, (size_t)N_NODES);
  unsigned* mx  = (unsigned*)carve(p, (size_t)N_NODES);
  float* xf     = carve(p, (size_t)N_SEG * XF_DIM);
  float* xf_gnn = carve(p, (size_t)N_SEG * H_DIM);
  float* y1     = carve(p, (size_t)N_SEG * H_DIM);
  float* y2     = carve(p, (size_t)N_SEG * H_DIM);

  // ---- pack all weight matrices into f16 WMMA fragment layout (tiny) ----
  _Float16* P_attr1 = pack_weight(p, attr1_W, A_DIM, stream);
  _Float16* P_attr2 = pack_weight(p, attr2_W, A_DIM, stream);
  _Float16* P_nlin  = pack_weight(p, node_lin_W, IN_DIM, stream);
  _Float16* P_Wl[2][3]; _Float16* P_Wr[2][3]; _Float16* P_We[2][3];
  for (int c = 0; c < 2; ++c)
    for (int l = 0; l < 3; ++l) {
      P_Wl[c][l] = pack_weight(p, c_Wl[c] + (size_t)l * H_DIM * H_DIM, H_DIM, stream);
      P_Wr[c][l] = pack_weight(p, c_Wr[c] + (size_t)l * H_DIM * H_DIM, H_DIM, stream);
      P_We[c][l] = pack_weight(p, c_We[c] + (size_t)l * H_DIM * H_DIM, H_DIM, stream);
    }
  _Float16* P_c3Wl = pack_weight(p, c3_Wl, XF_DIM, stream);
  _Float16* P_c3Wr = pack_weight(p, c3_Wr, XF_DIM, stream);
  _Float16* P_lin1 = pack_weight(p, lin1_W, XF_DIM, stream);
  _Float16* P_lin2 = pack_weight(p, lin2_W, H_DIM, stream);

  // ---- shared edge features: attr_f = attr @ attr1_W + attr1_b ----
  gemm_k16(attr, P_attr1, attr1_b, attr_f, N_EDGES, 0, stream);
  // nf0 = gelu(x_rec @ node_lin_W + node_lin_b)
  gemm_k32(x_rec, P_nlin, node_lin_b, nf0, N_NODES, 1, stream);

  // ---- conv1 / conv2 stacks (3 GATv2 layers each, residual to initial) ----
  const float* h_final[2];
  for (int c = 0; c < 2; ++c) {
    const float* pre = (c == 0) ? node_embed : nf0;
    const float* cur = pre;
    float* bufs[2] = {(c == 0) ? hA : nfA, (c == 0) ? hB : nfB};
    for (int l = 0; l < 3; ++l) {
      gemm_k64(cur,    P_Wl[c][l], c_bl[c] + l * H_DIM, xl, N_NODES, 0, stream);
      gemm_k64(cur,    P_Wr[c][l], c_br[c] + l * H_DIM, xr, N_NODES, 0, stream);
      gemm_k64(attr_f, P_We[c][l], nullptr,             ep, N_EDGES, 0, stream);
      run_gat_softmax_agg(xl, xr, ep, c_att[c] + l * H_DIM, src, dst,
                          logit, ex, mx, den, agg, N_EDGES, N_NODES, stream);
      float* nxt = bufs[l & 1];
      node_update<<<(N_NODES * H_DIM + 255) / 256, 256, 0, stream>>>(
          agg, c_bias[c] + l * H_DIM, pre, nxt, N_NODES);
      cur = nxt;
    }
    h_final[c] = cur;   // hA / nfA after 3 layers
  }

  // ---- xf = [seg_embed | x1e | x2e | x1n | x2n | t | w]  (S x 448) ----
  xf_init<<<(N_SEG * XF_DIM + 255) / 256, 256, 0, stream>>>(
      seg_embed, time_embed, week_embed, cur_t, cur_w, xf, N_SEG);
  spmm_scatter<<<(N_EDGES * H_DIM + 255) / 256, 256, 0, stream>>>(
      se_row, se_col, se_val, edge_embed, xf + 64, XF_DIM, N_EDGES);
  // x2e uses attr @ attr2_W + attr2_b — projected into the (free) ep buffer
  gemm_k16(attr, P_attr2, attr2_b, ep, N_EDGES, 0, stream);
  spmm_scatter<<<(N_EDGES * H_DIM + 255) / 256, 256, 0, stream>>>(
      se_row, se_col, se_val, ep, xf + 128, XF_DIM, N_EDGES);
  spmm_scatter<<<(N_EDGES * H_DIM + 255) / 256, 256, 0, stream>>>(
      sn_row, sn_col, sn_val, h_final[0], xf + 192, XF_DIM, N_EDGES);
  spmm_scatter<<<(N_EDGES * H_DIM + 255) / 256, 256, 0, stream>>>(
      sn_row, sn_col, sn_val, h_final[1], xf + 256, XF_DIM, N_EDGES);

  // ---- conv3: GATv2 over segment graph (edge_dim=1), no residual ----
  ep3_outer<<<(NNZ_SS * H_DIM + 255) / 256, 256, 0, stream>>>(ss_val, c3_We, ep, NNZ_SS);
  gemm_k448(xf, P_c3Wl, c3_bl, xl, N_SEG, 0, stream);
  gemm_k448(xf, P_c3Wr, c3_br, xr, N_SEG, 0, stream);
  run_gat_softmax_agg(xl, xr, ep, c3_att, ss_row, ss_col,
                      logit, ex, mx, den, agg, NNZ_SS, N_SEG, stream);
  node_update<<<(N_SEG * H_DIM + 255) / 256, 256, 0, stream>>>(
      agg, c3_bias, nullptr, xf_gnn, N_SEG);

  // ---- MLP head ----
  gemm_k448(xf, P_lin1, lin1_b, y1, N_SEG, 1, stream);
  gemm_k64(y1, P_lin2, lin2_b, y2, N_SEG, 1, stream);
  head_out<<<(N_SEG + 255) / 256, 256, 0, stream>>>(y2, xf_gnn, out_W, out_b, out, N_SEG);
}
